// HoughVoting_89060441850445
// MI455X (gfx1250) — matrix operand order repaired and checked by reading, not compiled, not run on hardware
//
#include <hip/hip_runtime.h>
#include <hip/hip_bf16.h>
#include <math.h>

typedef __attribute__((ext_vector_type(16))) _Float16 v16h;
typedef __attribute__((ext_vector_type(8)))  float    v8f;

#define HOUGH_W   160
#define HOUGH_H   120
#define HOUGH_HW  19200      // H*W
#define SKIP      8
#define NPIX      2400       // HW / SKIP
#define NCHUNK    75         // NPIX / 32
#define NCLS      8
#define THR       0.9f
#define EPSV      1e-6f

// Raw v_sqrt_f32: inputs here are exact integers in [0, 51200] (never denormal),
// so the precise-sqrt denormal-rescue sequence sqrtf() emits is dead weight.
__device__ __forceinline__ float fast_sqrtf(float x) {
    return __builtin_amdgcn_sqrtf(x);
}

// ---------------------------------------------------------------------------
// Kernel 0: zero the argmax keys and per-class valid-pixel counters.
// ---------------------------------------------------------------------------
__global__ void hough_init(unsigned long long* __restrict__ keys,
                           int* __restrict__ nvalid) {
    int t = threadIdx.x;
    if (t < NCLS) { keys[t] = 0ull; nvalid[t] = 0; }
}

// ---------------------------------------------------------------------------
// Kernel 1: per-voting-pixel tables.
// pixA: {nx, ny, xs, ys}  (nx=ny=0 when invalid -> inlier test auto-fails)
// pixB: {depth, label_f}  (label_f = -1 when invalid)
// Also counts valid pixels per class.
// ---------------------------------------------------------------------------
__global__ void hough_prep(const int* __restrict__ labels,
                           const int* __restrict__ masks,
                           const float* __restrict__ vp,    // [3C, H, W]
                           float4* __restrict__ pixA,
                           float2* __restrict__ pixB,
                           int* __restrict__ nvalid) {
    int p = blockIdx.x * blockDim.x + threadIdx.x;
    if (p >= NPIX) return;
    int idx = p * SKIP;
    int lab = labels[idx];
    int msk = masks[idx];
    bool valid = (msk > 0) && (lab > 0);
    float vx = vp[(lab * 3 + 0) * HOUGH_HW + idx];
    float vy = vp[(lab * 3 + 1) * HOUGH_HW + idx];
    float vz = vp[(lab * 3 + 2) * HOUGH_HW + idx];
    float nrm = sqrtf(vx * vx + vy * vy) + EPSV;         // keep precise (matches ref)
    float4 a;
    a.x = valid ? (vx / nrm) : 0.0f;
    a.y = valid ? (vy / nrm) : 0.0f;
    a.z = (float)((p % 20) * SKIP);                      // xs = (p*8) % 160
    a.w = (float)(p / 20);                               // ys = (p*8) / 160
    pixA[p] = a;
    float2 b;
    b.x = expf(vz);
    b.y = valid ? (float)lab : -1.0f;
    pixB[p] = b;
    if (valid) atomicAdd(&nvalid[lab], 1);
}

// ---------------------------------------------------------------------------
// Kernel 1b: materialize the loop-invariant WMMA B-matrix tiles, already in
// per-lane register layout.  Btab[chunk][lane][j] (f16), where for lane n:
//   col = n&15, K = ((n>=16)?16:0) + j, pixel k = chunk*32 + K
//   value = col<8 ? onehot(label==col) : onehot(label==col-8) * depth
// 75 chunks * 32 lanes * 16 halves = 38400 f16 = 75 KB (L2-resident).
// ---------------------------------------------------------------------------
__global__ void hough_prep_b(const float2* __restrict__ pixB,
                             _Float16* __restrict__ Btab) {
    int t = blockIdx.x * blockDim.x + threadIdx.x;
    if (t >= NCHUNK * 512) return;
    int chunk = t >> 9;                                  // /512
    int r     = t & 511;
    int ln    = r >> 4;                                  // lane 0..31
    int j     = r & 15;
    int col   = ln & 15;
    int k     = chunk * 32 + ((ln >= 16) ? 16 : 0) + j;
    float2 db = pixB[k];
    int lab = (int)db.y;                                 // -1 when invalid
    float v;
    if (col < NCLS) v = (lab == col) ? 1.0f : 0.0f;
    else            v = (lab == col - NCLS) ? db.x : 0.0f;
    Btab[t] = (_Float16)v;
}

// ---------------------------------------------------------------------------
// Kernel 2: WMMA Hough accumulation.
// Each wave owns 16 Hough cells (M), sweeps 2400 pixels in K=32 chunks.
// A (16x32 f16): inlier flags computed on the fly from the LDS pixel table.
// B (32x16 f16): loaded pre-packed from Btab (2 x b128 per chunk).
// Two accumulators break the WMMA->WMMA RAW chain on C.
// Winner: 64-bit atomicMax key = (votes<<32)|(HW-1-loc)  (first-max ties).
// ---------------------------------------------------------------------------
__device__ __forceinline__ v8f hough_step(const float4* s_pix,
                                          const _Float16* __restrict__ Btab,
                                          int ch, int lane, int aOfs,
                                          float gx, float gy, v8f acc) {
    v16h a;
#pragma unroll
    for (int j = 0; j < 16; ++j) {
        // K = 2*aOfs + j (contiguous, matches Btab packing)
        int k = ch * 32 + aOfs * 2 + j;
        float4 q = s_pix[k];
        float dx = gx - q.z;
        float dy = gy - q.w;
        float dt = fmaf(dy, q.y, dx * q.x);              // dot(ray, dir)
        float d2 = fmaf(dy, dy, dx * dx);                // exact integer >= 0
        float ds = fast_sqrtf(d2);
        a[j] = (dt > THR * ds) ? (_Float16)1.0f : (_Float16)0.0f;
    }
    v16h b = *(const v16h*)(Btab + (ch * 32 + lane) * 16);
    return __builtin_amdgcn_wmma_f32_16x16x32_f16(
        false, a, false, b, (short)0, acc, false, false);
}

__global__ __launch_bounds__(256)
void hough_vote_wmma(const float4* __restrict__ pixA,
                     const _Float16* __restrict__ Btab,
                     unsigned long long* __restrict__ keys) {
    __shared__ float4 s_pix[NPIX];                       // 38400 B
    __builtin_prefetch(Btab, 0, 1);
    for (int i = threadIdx.x; i < NPIX; i += 256) s_pix[i] = pixA[i];
    __syncthreads();

    const int lane    = threadIdx.x & 31;
    const int wave    = threadIdx.x >> 5;
    const int baseLoc = (blockIdx.x * 8 + wave) * 16;    // 16 Hough cells/wave
    const int hiHalf  = (lane >= 16) ? 1 : 0;
    const int col     = lane & 15;                       // D column for this lane
    const int aOfs    = hiHalf ? 8 : 0;                  // A K-offset = 2*aOfs (contiguous order)

    const int locA = baseLoc + (lane & 15);              // A row for this lane
    const float gx = (float)(locA % HOUGH_W);
    const float gy = (float)(locA / HOUGH_W);

    v8f acc0 = {}, acc1 = {};
    for (int ch = 0; ch < NCHUNK - 1; ch += 2) {
        acc0 = hough_step(s_pix, Btab, ch,     lane, aOfs, gx, gy, acc0);
        acc1 = hough_step(s_pix, Btab, ch + 1, lane, aOfs, gx, gy, acc1);
    }
    acc0 = hough_step(s_pix, Btab, NCHUNK - 1, lane, aOfs, gx, gy, acc0);

    // D layout: VGPR r holds (M = r + 8*hiHalf, N = col)
    if (col < NCLS) {
#pragma unroll
        for (int r = 0; r < 8; ++r) {
            int loc = baseLoc + r + (hiHalf ? 8 : 0);
            float votes = acc0[r] + acc1[r];             // exact integer in f32
            int v = (int)(votes + 0.5f);
            unsigned long long key =
                ((unsigned long long)(unsigned int)v << 32) |
                (unsigned int)(HOUGH_HW - 1 - loc);
            atomicMax(&keys[col], key);
        }
    }
}

// ---------------------------------------------------------------------------
// Kernel 3: per class, recompute depth sum at the winning center in f32,
// then emit top_box[8,7] and top_pose[8,7].
// ---------------------------------------------------------------------------
__global__ __launch_bounds__(256)
void hough_finalize(const float4* __restrict__ pixA,
                    const float2* __restrict__ pixB,
                    const unsigned long long* __restrict__ keys,
                    const int* __restrict__ nvalid,
                    const float* __restrict__ extents,   // [C,3]
                    const float* __restrict__ poses,     // [C,7] (B=1)
                    const float* __restrict__ meta,      // [9]
                    float* __restrict__ out) {
    const int c = blockIdx.x;
    unsigned long long key = keys[c];
    int vmax = (int)(key >> 32);
    int best = HOUGH_HW - 1 - (int)(key & 0xffffffffu);
    float gx = (float)(best % HOUGH_W);
    float gy = (float)(best / HOUGH_W);

    __shared__ float sred[256];
    float sum = 0.0f;
    for (int p = threadIdx.x; p < NPIX; p += 256) {
        float2 db = pixB[p];
        int lab = (int)db.y;                             // only valid pixels match
        if (lab == c) {
            float4 q = pixA[p];
            float dx = gx - q.z;
            float dy = gy - q.w;
            float dt = fmaf(dy, q.y, dx * q.x);
            float ds = fast_sqrtf(fmaf(dy, dy, dx * dx));
            if (dt > THR * ds) sum += db.x;
        }
    }
    sred[threadIdx.x] = sum;
    __syncthreads();
    for (int s = 128; s > 0; s >>= 1) {
        if ((int)threadIdx.x < s) sred[threadIdx.x] += sred[threadIdx.x + s];
        __syncthreads();
    }

    if (threadIdx.x == 0) {
        float dsum   = sred[0];
        float vmax_f = (float)vmax;
        float dbar   = dsum / fmaxf(vmax_f, 1.0f);
        float fx = meta[0], pxc = meta[2], fy = meta[4], pyc = meta[5];
        float ex = extents[c * 3 + 0];
        float ey = extents[c * 3 + 1];
        float ez = extents[c * 3 + 2];
        float diag   = sqrtf(ex * ex + ey * ey + ez * ez);
        float safe_d = fmaxf(dbar, EPSV);
        float bw = diag * fx / safe_d;
        float bh = diag * fy / safe_d;
        float score = vmax_f / fmaxf((float)nvalid[c], 1.0f);

        float* box = out + c * 7;                        // top_box rows first
        box[0] = 0.0f;
        box[1] = (float)c;
        box[2] = gx - bw * 0.5f;
        box[3] = gy - bh * 0.5f;
        box[4] = gx + bw * 0.5f;
        box[5] = gy + bh * 0.5f;
        box[6] = score;

        float* tp = out + NCLS * 7 + c * 7;              // then top_pose rows
        tp[0] = poses[c * 7 + 0];
        tp[1] = poses[c * 7 + 1];
        tp[2] = poses[c * 7 + 2];
        tp[3] = poses[c * 7 + 3];
        tp[4] = (gx - pxc) * dbar / fmaxf(fx, EPSV);
        tp[5] = (gy - pyc) * dbar / fmaxf(fy, EPSV);
        tp[6] = dbar;
    }
}

// ---------------------------------------------------------------------------
extern "C" void kernel_launch(void* const* d_in, const int* in_sizes, int n_in,
                              void* d_out, int out_size, void* d_ws, size_t ws_size,
                              hipStream_t stream) {
    const int*   labels  = (const int*)d_in[0];
    const int*   masks   = (const int*)d_in[1];
    const float* vp      = (const float*)d_in[2];
    const float* extents = (const float*)d_in[3];
    const float* poses   = (const float*)d_in[4];
    const float* meta    = (const float*)d_in[5];
    float* out = (float*)d_out;

    char* ws = (char*)d_ws;
    float4*             pixA   = (float4*)(ws + 0);                  //  38400 B
    float2*             pixB   = (float2*)(ws + 38400);              //  19200 B
    _Float16*           Btab   = (_Float16*)(ws + 57600);            //  76800 B
    unsigned long long* keys   = (unsigned long long*)(ws + 134400); //     64 B
    int*                nvalid = (int*)(ws + 134464);                //     32 B

    hough_init<<<1, 32, 0, stream>>>(keys, nvalid);
    hough_prep<<<(NPIX + 255) / 256, 256, 0, stream>>>(labels, masks, vp,
                                                       pixA, pixB, nvalid);
    hough_prep_b<<<(NCHUNK * 512 + 255) / 256, 256, 0, stream>>>(pixB, Btab);
    hough_vote_wmma<<<HOUGH_HW / 128, 256, 0, stream>>>(pixA, Btab, keys);
    hough_finalize<<<NCLS, 256, 0, stream>>>(pixA, pixB, keys, nvalid,
                                             extents, poses, meta, out);
}
